// TimeSeriesModel_NStep_Combined_29463475650708
// MI455X (gfx1250) — matrix operand order repaired and checked by reading, not compiled, not run on hardware
//
#include <hip/hip_runtime.h>
#include <hip/hip_bf16.h>

typedef __attribute__((ext_vector_type(16))) _Float16 v16h;
typedef __attribute__((ext_vector_type(8)))  _Float16 v8h;
typedef __attribute__((ext_vector_type(8)))  float    v8f;

#define B_SZ 256
#define L_SZ 512
#define D_SZ 64
#define H_SZ 1024
#define S_SZ 24
#define NWG  64
#define TPB  256

// LDS pitches (halves). 1032*2B = 2064B = 516 dwords; 516 % 64 == 4 -> the 16
// fragment columns hit banks 4n..4n+3: all 64 banks exactly once per half-wave.
#define PW 1032
#define PI 72
#define LDS_HALVES (48 * PW + 48 * PI)

// ---------------------------------------------------------------- helpers

__device__ __forceinline__ v8f splat8(float x){
  v8f v;
  for (int i = 0; i < 8; ++i) v[i] = x;
  return v;
}

__device__ __forceinline__ v16h cat16(v8h a, v8h b){
  return __builtin_shufflevector(a, b, 0,1,2,3,4,5,6,7,8,9,10,11,12,13,14,15);
}

// A fragment (16x32 f16, M x K) from global. p0 at (row0, kbase); ld in halves.
__device__ __forceinline__ v16h load_fragA(const _Float16* p0, size_t ld, int lane){
  const _Float16* p = p0 + (size_t)(lane & 15) * ld + ((lane >> 4) * 16);
  v8h a0 = *(const v8h*)p;
  v8h a1 = *(const v8h*)(p + 8);
  return cat16(a0, a1);
}

// B fragment (32x16 f16, K x N), B[k][n] = W[n][k], from global. p0 at (nrow0, kbase).
__device__ __forceinline__ v16h load_fragB(const _Float16* p0, size_t ld, int lane){
  const _Float16* p = p0 + (size_t)(lane & 15) * ld + ((lane >> 4) * 16);
  v8h b0 = *(const v8h*)p;
  v8h b1 = *(const v8h*)(p + 8);
  return cat16(b0, b1);
}

// B fragment from an LDS-staged weight slice (pitch in halves).
__device__ __forceinline__ v16h lds_fragB(const _Float16* s, int pitch, int lane){
  const _Float16* p = s + (lane & 15) * pitch + ((lane >> 4) * 16);
  v8h b0 = *(const v8h*)p;
  v8h b1 = *(const v8h*)(p + 8);
  return cat16(b0, b1);
}

__device__ __forceinline__ v8f wmma16(v16h a, v16h b, v8f c){
  return __builtin_amdgcn_wmma_f32_16x16x32_f16(false, a, false, b, (short)0, c, false, false);
}

__device__ __forceinline__ float sigm(float x){ return 1.0f / (1.0f + __expf(-x)); }

// Cooperatively copy one 16-row weight block (kels columns) into LDS with padding.
__device__ __forceinline__ void fill_lds(_Float16* s, const _Float16* W, int kels,
                                         size_t ldW, int pitch){
  const int cpr = kels >> 3;              // v8h chunks per row
  const int total = 16 * cpr;
  for (int j = threadIdx.x; j < total; j += TPB){
    int r  = j / cpr;
    int kc = (j - r * cpr) << 3;
    *(v8h*)&s[r * pitch + kc] = *(const v8h*)&W[(size_t)r * ldW + kc];
  }
}

// Agent-scope sense-reversing grid barrier. bar[0]=count, bar[1]=generation.
__device__ __forceinline__ void grid_barrier(unsigned* bar){
  __threadfence();
  __syncthreads();
  if (threadIdx.x == 0){
    unsigned g = __hip_atomic_load(&bar[1], __ATOMIC_RELAXED, __HIP_MEMORY_SCOPE_AGENT);
    unsigned a = __hip_atomic_fetch_add(&bar[0], 1u, __ATOMIC_ACQ_REL, __HIP_MEMORY_SCOPE_AGENT);
    if (a == (unsigned)(NWG - 1)){
      __hip_atomic_store(&bar[0], 0u, __ATOMIC_RELAXED, __HIP_MEMORY_SCOPE_AGENT);
      __hip_atomic_store(&bar[1], g + 1u, __ATOMIC_RELEASE, __HIP_MEMORY_SCOPE_AGENT);
    } else {
      while (__hip_atomic_load(&bar[1], __ATOMIC_ACQUIRE, __HIP_MEMORY_SCOPE_AGENT) == g)
        __builtin_amdgcn_s_sleep(1);
    }
  }
  __syncthreads();
}

// ---------------------------------------------------------------- kernels

__global__ void f32_to_f16(const float* __restrict__ src, _Float16* __restrict__ dst, int n){
  int i = blockIdx.x * blockDim.x + threadIdx.x;
  if (i < n) dst[i] = (_Float16)src[i];
}

// double exponential smoothing: x -> x2 (f16), record st11/st12 at COLS {1,2}
__global__ void es_kernel(const float* __restrict__ x, _Float16* __restrict__ x2h,
                          float* __restrict__ st11, float* __restrict__ st12){
  int tid = blockIdx.x * blockDim.x + threadIdx.x;   // 16384 = B*D
  int b = tid >> 6, d = tid & 63;
  const float* xp = x + ((size_t)b * L_SZ) * D_SZ + d;
  float p1 = xp[0];
  float p2 = p1;
  x2h[((size_t)b * L_SZ) * D_SZ + d] = (_Float16)p2;
  for (int t = 1; t < L_SZ; ++t){
    float xv = xp[(size_t)t * D_SZ];
    p1 = 0.7f * xv + 0.3f * p1;   // alpha = 0.3
    p2 = 0.5f * p1 + 0.5f * p2;   // beta  = 0.5
    x2h[((size_t)b * L_SZ + t) * D_SZ + d] = (_Float16)p2;
  }
  if (d == 1 || d == 2){ st11[b * 2 + (d - 1)] = p1; st12[b * 2 + (d - 1)] = p2; }
}

// persistent encoder + fci/dec-input GEMM + decoder
__global__ __launch_bounds__(TPB) void gru_persistent(
    const _Float16* __restrict__ x2h,
    const _Float16* __restrict__ eWih, const _Float16* __restrict__ eWhh,
    const float* __restrict__ ebih, const float* __restrict__ ebhh,
    const _Float16* __restrict__ dWih, const _Float16* __restrict__ dWhh,
    const float* __restrict__ dbih, const float* __restrict__ dbhh,
    const _Float16* __restrict__ fciW, const float* __restrict__ fcib,
    _Float16* __restrict__ hbuf,          // 2 * B * H halves (ping-pong)
    _Float16* __restrict__ dechs, unsigned* __restrict__ bar)
{
  extern __shared__ _Float16 smem[];
  _Float16* sWhh = smem;                  // 48 rows x PW
  _Float16* sWih = smem + 48 * PW;        // 48 rows x PI

  const int lane = threadIdx.x & 31;
  const int wave = threadIdx.x >> 5;
  const int ncol = lane & 15;
  const int rsel = lane >> 4;
  const int nb   = blockIdx.x * 16;       // this WG's 16-column slice of h
  const int col  = nb + ncol;
  int mrow[2] = { (wave * 2) * 16, (wave * 2 + 1) * 16 };

  // loop-invariant per-lane bias values (depend only on column)
  const float ebr = ebih[col]            + ebhh[col];
  const float ebz = ebih[H_SZ + col]     + ebhh[H_SZ + col];
  const float ebi = ebih[2 * H_SZ + col];
  const float ebh = ebhh[2 * H_SZ + col];
  const float gbr = dbih[col];
  const float gbz = dbih[H_SZ + col];
  const float gbi = dbih[2 * H_SZ + col];
  const float dbr = dbhh[col];
  const float dbz = dbhh[H_SZ + col];
  const float dbh = dbhh[2 * H_SZ + col];
  const float fbv = fcib[col];

  // stage this WG's encoder weight slices into LDS (reused for all 512 steps)
  for (int g = 0; g < 3; ++g){
    fill_lds(sWhh + g * 16 * PW, eWhh + (size_t)(g * H_SZ + nb) * H_SZ, H_SZ, H_SZ, PW);
    fill_lds(sWih + g * 16 * PI, eWih + (size_t)(g * H_SZ + nb) * D_SZ, D_SZ, D_SZ, PI);
  }
  __syncthreads();

  v8f hreg[2]; hreg[0] = splat8(0.f); hreg[1] = splat8(0.f);   // h0 = 0

  // ---------------- encoder: 512 steps ----------------
  #pragma unroll 1
  for (int t = 0; t < L_SZ; ++t){
    const _Float16* hc = hbuf + (size_t)(t & 1) * (B_SZ * H_SZ);
    _Float16*       hn = hbuf + (size_t)((t + 1) & 1) * (B_SZ * H_SZ);
    v8f aR[2], aZ[2], aI[2], aH[2];
    for (int m = 0; m < 2; ++m){
      aR[m] = splat8(ebr); aZ[m] = splat8(ebz);
      aI[m] = splat8(ebi); aH[m] = splat8(ebh);
    }
    // input projection: x2[:,t,:] @ enc_Wih^T  (K = 64), B from LDS
    #pragma unroll 1
    for (int kk = 0; kk < D_SZ; kk += 32){
      v16h bR = lds_fragB(sWih +  0 * PI + kk, PI, lane);
      v16h bZ = lds_fragB(sWih + 16 * PI + kk, PI, lane);
      v16h bI = lds_fragB(sWih + 32 * PI + kk, PI, lane);
      for (int m = 0; m < 2; ++m){
        v16h a = load_fragA(x2h + ((size_t)mrow[m] * L_SZ + t) * D_SZ + kk,
                            (size_t)L_SZ * D_SZ, lane);
        aR[m] = wmma16(a, bR, aR[m]);
        aZ[m] = wmma16(a, bZ, aZ[m]);
        aI[m] = wmma16(a, bI, aI[m]);
      }
    }
    // recurrent projection: h @ enc_Whh^T  (K = 1024), B from LDS
    #pragma unroll 1
    for (int kk = 0; kk < H_SZ; kk += 32){
      v16h bR = lds_fragB(sWhh +  0 * PW + kk, PW, lane);
      v16h bZ = lds_fragB(sWhh + 16 * PW + kk, PW, lane);
      v16h bH = lds_fragB(sWhh + 32 * PW + kk, PW, lane);
      for (int m = 0; m < 2; ++m){
        v16h a = load_fragA(hc + (size_t)mrow[m] * H_SZ + kk, H_SZ, lane);
        aR[m] = wmma16(a, bR, aR[m]);
        aZ[m] = wmma16(a, bZ, aZ[m]);
        aH[m] = wmma16(a, bH, aH[m]);
      }
    }
    // GRU gate epilogue
    for (int m = 0; m < 2; ++m){
      v8f hnew;
      for (int i = 0; i < 8; ++i){
        float r = sigm(aR[m][i]);
        float z = sigm(aZ[m][i]);
        float n = tanhf(aI[m][i] + r * aH[m][i]);
        float h = (1.f - z) * n + z * hreg[m][i];
        hnew[i] = h;
        hn[(size_t)(mrow[m] + i + rsel * 8) * H_SZ + col] = (_Float16)h;
      }
      hreg[m] = hnew;
    }
    grid_barrier(bar);
  }

  // ---------------- mid: h0d = hT@fci^T + b ; gi_dec = hT@dWih^T + bih (into regs)
  const _Float16* hT = hbuf;              // parity 0 after 512 steps
  _Float16*       h1 = hbuf + (size_t)(B_SZ * H_SZ);
  v8f giR[2], giZ[2], giI[2], hd[2];
  for (int m = 0; m < 2; ++m){
    giR[m] = splat8(gbr); giZ[m] = splat8(gbz); giI[m] = splat8(gbi); hd[m] = splat8(fbv);
  }
  #pragma unroll 1
  for (int kk = 0; kk < H_SZ; kk += 32){
    v16h bR = load_fragB(dWih + (size_t)(0 * H_SZ + nb) * H_SZ + kk, H_SZ, lane);
    v16h bZ = load_fragB(dWih + (size_t)(1 * H_SZ + nb) * H_SZ + kk, H_SZ, lane);
    v16h bI = load_fragB(dWih + (size_t)(2 * H_SZ + nb) * H_SZ + kk, H_SZ, lane);
    v16h bF = load_fragB(fciW + (size_t)nb * H_SZ + kk, H_SZ, lane);
    for (int m = 0; m < 2; ++m){
      v16h a = load_fragA(hT + (size_t)mrow[m] * H_SZ + kk, H_SZ, lane);
      giR[m] = wmma16(a, bR, giR[m]);
      giZ[m] = wmma16(a, bZ, giZ[m]);
      giI[m] = wmma16(a, bI, giI[m]);
      hd[m]  = wmma16(a, bF, hd[m]);
    }
  }
  for (int m = 0; m < 2; ++m){
    for (int i = 0; i < 8; ++i)
      h1[(size_t)(mrow[m] + i + rsel * 8) * H_SZ + col] = (_Float16)hd[m][i];
    hreg[m] = hd[m];    // decoder h0 = h0d
  }
  grid_barrier(bar);

  // re-stage LDS with decoder recurrent weights (encoder slice no longer needed)
  for (int g = 0; g < 3; ++g)
    fill_lds(sWhh + g * 16 * PW, dWhh + (size_t)(g * H_SZ + nb) * H_SZ, H_SZ, H_SZ, PW);
  __syncthreads();

  // ---------------- decoder: 24 steps (constant input held in giR/giZ/giI)
  #pragma unroll 1
  for (int s = 0; s < S_SZ; ++s){
    const _Float16* hc = hbuf + (size_t)((s + 1) & 1) * (B_SZ * H_SZ);
    _Float16*       hn = hbuf + (size_t)(s & 1) * (B_SZ * H_SZ);
    v8f aR[2], aZ[2], aH[2];
    for (int m = 0; m < 2; ++m){
      for (int i = 0; i < 8; ++i){ aR[m][i] = giR[m][i] + dbr; aZ[m][i] = giZ[m][i] + dbz; }
      aH[m] = splat8(dbh);
    }
    #pragma unroll 1
    for (int kk = 0; kk < H_SZ; kk += 32){
      v16h bR = lds_fragB(sWhh +  0 * PW + kk, PW, lane);
      v16h bZ = lds_fragB(sWhh + 16 * PW + kk, PW, lane);
      v16h bH = lds_fragB(sWhh + 32 * PW + kk, PW, lane);
      for (int m = 0; m < 2; ++m){
        v16h a = load_fragA(hc + (size_t)mrow[m] * H_SZ + kk, H_SZ, lane);
        aR[m] = wmma16(a, bR, aR[m]);
        aZ[m] = wmma16(a, bZ, aZ[m]);
        aH[m] = wmma16(a, bH, aH[m]);
      }
    }
    for (int m = 0; m < 2; ++m){
      v8f hnew;
      for (int i = 0; i < 8; ++i){
        float r = sigm(aR[m][i]);
        float z = sigm(aZ[m][i]);
        float n = tanhf(giI[m][i] + r * aH[m][i]);
        float h = (1.f - z) * n + z * hreg[m][i];
        hnew[i] = h;
        int row = mrow[m] + i + rsel * 8;
        hn[(size_t)row * H_SZ + col] = (_Float16)h;
        dechs[((size_t)s * B_SZ + row) * H_SZ + col] = (_Float16)h;
      }
      hreg[m] = hnew;
    }
    grid_barrier(bar);
  }
}

// fc projection + sequential de-smoothing; one thread per (batch, class)
__global__ void final_fc(const _Float16* __restrict__ dechs,
                         const float* __restrict__ fcW, const float* __restrict__ fcb,
                         const float* __restrict__ st11, const float* __restrict__ st12,
                         float* __restrict__ out){
  int tid = blockIdx.x * blockDim.x + threadIdx.x;   // 512 = B*NC
  if (tid >= B_SZ * 2) return;
  int b = tid >> 1, c = tid & 1;
  float prev2 = st12[b * 2 + c];
  float prev1 = st11[b * 2 + c];
  const float* w = fcW + (size_t)c * H_SZ;
  for (int t = 0; t < S_SZ; ++t){
    const _Float16* hrow = dechs + ((size_t)t * B_SZ + b) * H_SZ;
    float raw = fcb[c];
    for (int j = 0; j < H_SZ; j += 8){
      v8h hv = *(const v8h*)&hrow[j];
      for (int e = 0; e < 8; ++e) raw += (float)hv[e] * w[j + e];
    }
    float o1 = (raw - 0.5f * prev2) / 0.5f;   // beta de-smoothing
    float o2 = (o1 - 0.3f * prev1) / 0.7f;    // alpha de-smoothing
    prev2 = raw;
    prev1 = o1;
    out[(size_t)b * (S_SZ * 2) + t * 2 + c] = o2;
  }
}

// ---------------------------------------------------------------- launch

extern "C" void kernel_launch(void* const* d_in, const int* in_sizes, int n_in,
                              void* d_out, int out_size, void* d_ws, size_t ws_size,
                              hipStream_t stream) {
  (void)in_sizes; (void)n_in; (void)out_size; (void)ws_size;

  const float* x      = (const float*)d_in[0];
  const float* eWih_f = (const float*)d_in[1];
  const float* eWhh_f = (const float*)d_in[2];
  const float* ebih   = (const float*)d_in[3];
  const float* ebhh   = (const float*)d_in[4];
  const float* dWih_f = (const float*)d_in[5];
  const float* dWhh_f = (const float*)d_in[6];
  const float* dbih   = (const float*)d_in[7];
  const float* dbhh   = (const float*)d_in[8];
  const float* fciW_f = (const float*)d_in[9];
  const float* fcib   = (const float*)d_in[10];
  const float* fcW    = (const float*)d_in[11];
  const float* fcb    = (const float*)d_in[12];

  char* base = (char*)d_ws;
  size_t off = 0;
  auto alloc = [&](size_t bytes) -> char* {
    char* r = base + off;
    off = (off + bytes + 255) & ~(size_t)255;
    return r;
  };
  _Float16* x2h   = (_Float16*)alloc((size_t)B_SZ * L_SZ * D_SZ * 2);
  _Float16* eWih  = (_Float16*)alloc((size_t)3 * H_SZ * D_SZ * 2);
  _Float16* eWhh  = (_Float16*)alloc((size_t)3 * H_SZ * H_SZ * 2);
  _Float16* dWih  = (_Float16*)alloc((size_t)3 * H_SZ * H_SZ * 2);
  _Float16* dWhh  = (_Float16*)alloc((size_t)3 * H_SZ * H_SZ * 2);
  _Float16* fciW  = (_Float16*)alloc((size_t)H_SZ * H_SZ * 2);
  _Float16* hbuf  = (_Float16*)alloc((size_t)2 * B_SZ * H_SZ * 2);
  _Float16* dechs = (_Float16*)alloc((size_t)S_SZ * B_SZ * H_SZ * 2);
  float*    st11  = (float*)alloc((size_t)B_SZ * 2 * 4);
  float*    st12  = (float*)alloc((size_t)B_SZ * 2 * 4);
  unsigned* bar   = (unsigned*)alloc(256);

  // zero h parity-0 (encoder initial state) and the grid barrier
  hipMemsetAsync(hbuf, 0, (size_t)B_SZ * H_SZ * 2, stream);
  hipMemsetAsync(bar, 0, 8, stream);

  auto cvt = [&](const float* s, _Float16* d, int n){
    f32_to_f16<<<(n + TPB - 1) / TPB, TPB, 0, stream>>>(s, d, n);
  };
  cvt(eWih_f, eWih, 3 * H_SZ * D_SZ);
  cvt(eWhh_f, eWhh, 3 * H_SZ * H_SZ);
  cvt(dWih_f, dWih, 3 * H_SZ * H_SZ);
  cvt(dWhh_f, dWhh, 3 * H_SZ * H_SZ);
  cvt(fciW_f, fciW, H_SZ * H_SZ);

  es_kernel<<<(B_SZ * D_SZ) / TPB, TPB, 0, stream>>>(x, x2h, st11, st12);

  size_t ldsBytes = (size_t)LDS_HALVES * 2;   // ~103.5 KB of the 320 KB WGP LDS
  gru_persistent<<<NWG, TPB, ldsBytes, stream>>>(x2h, eWih, eWhh, ebih, ebhh,
                                                 dWih, dWhh, dbih, dbhh,
                                                 fciW, fcib, hbuf, dechs, bar);

  final_fc<<<2, TPB, 0, stream>>>(dechs, fcW, fcb, st11, st12, (float*)d_out);
}
